// ProvenanceAttention_41609643164162
// MI455X (gfx1250) — matrix operand order repaired and checked by reading, not compiled, not run on hardware
//
#include <hip/hip_runtime.h>
#include <hip/hip_bf16.h>

// ---------------------------------------------------------------------------
// ProvenanceAttention for MI455X (gfx1250, wave32, WMMA).
// All six GEMMs run through v_wmma_f32_16x16x32_bf16 with fp32 accumulators.
// Weights are pre-transposed/converted to bf16 [n][k] so every B-fragment is
// two contiguous global_load_b128; activations are pre-converted to bf16 so
// A-fragments are contiguous b128 loads too. Projection waves compute 16x64
// output tiles (4 WMMAs per A-fragment). The attention core keeps the full
// 16x4096 fp32 score panel in LDS (256KB of MI455X's 320KB/WGP), giving exact
// softmax weights for both the provenance output (head-mean, f32 HW atomics)
// and the PV WMMA chain in a single pass.
// ---------------------------------------------------------------------------

typedef __attribute__((ext_vector_type(16))) __bf16 v16bf;
typedef __attribute__((ext_vector_type(8)))  __bf16 v8bf;
typedef __attribute__((ext_vector_type(8)))  float  v8f;

#define HIDDEN   1024
#define HEADS    16
#define HEAD_DIM 64
#define BB       4
#define TT       512
#define SS       4096

// -------- fragment loaders (contiguous k = 16*hlf + j for A and B, wave32) ----------

// A-matrix 16x32 fragment from a bf16 row-major source (one 32B run).
// lane 0-15: M=lane; element j -> K = hlf*16+j.
__device__ __forceinline__ v16bf load_a_from_bf16(const __bf16* row,
                                                  int kbase, int hlf) {
  return *(const v16bf*)(row + kbase + hlf * 16);
}

// B-matrix 32x16 fragment from a bf16 K-contiguous (transposed) source:
// Wt[n][k].  lane: N = lane&15 ; element j -> K = 16*(lane>>4) + j, so each
// lane reads 16 contiguous bf16 (one 32B run -> two b128 loads).
__device__ __forceinline__ v16bf load_b_from_bf16t(const __bf16* Wt, int ld,
                                                   int kbase, int n0,
                                                   int lane) {
  int n = lane & 15, hlf = lane >> 4;
  return *(const v16bf*)(Wt + (size_t)(n0 + n) * ld + kbase + hlf * 16);
}

// -------- init / conversion helpers -----------------------------------------

__global__ void zero_f32_kernel(float* p, int n) {
  int i = blockIdx.x * blockDim.x + threadIdx.x;
  if (i < n) p[i] = 0.0f;
}

__global__ void cvt_f32_to_bf16_kernel(const float* __restrict__ src,
                                       __bf16* __restrict__ dst, int n) {
  int i = blockIdx.x * blockDim.x + threadIdx.x;
  if (i < n) dst[i] = (__bf16)src[i];
}

// Wt[n][k] = bf16(W[k][n]); LDS-tiled 32x32 transpose, 256 threads/block.
__global__ void __launch_bounds__(256)
transpose_cvt_kernel(const float* __restrict__ W, __bf16* __restrict__ Wt) {
  __shared__ float t[32][33];
  int tx = threadIdx.x & 31, ty = threadIdx.x >> 5;  // ty in 0..7
  int nt = blockIdx.x & 31, kt = blockIdx.x >> 5;    // 32x32 tile grid
  int n0 = nt * 32, k0 = kt * 32;
#pragma unroll
  for (int i = 0; i < 4; ++i)
    t[ty + i * 8][tx] = W[(size_t)(k0 + ty + i * 8) * HIDDEN + n0 + tx];
  __syncthreads();
#pragma unroll
  for (int i = 0; i < 4; ++i)
    Wt[(size_t)(n0 + ty + i * 8) * HIDDEN + k0 + tx] =
        (__bf16)t[tx][ty + i * 8];
}

// -------- projection GEMM: out = X @ W + bias (+seg_emb), bf16 output -------
// One wave computes a 16x64 tile: 4 accumulators share each A-fragment.
// mode 0: Q  -> row-major [M][1024]
// mode 1: K  -> [b][h][s][d]  (+ bias + seg_emb[segment_ids])
// mode 2: V  -> [b][h][d][s]  (transposed so PV B-fragments are contiguous)

__global__ void __launch_bounds__(128)
proj_kernel(const __bf16* __restrict__ X, const __bf16* __restrict__ Wt,
            const float* __restrict__ bias, const int* __restrict__ seg_ids,
            const float* __restrict__ seg_emb, __bf16* __restrict__ out,
            int Mrows, int rowsPerB, int mode) {
  int waveId = blockIdx.x * 4 + (threadIdx.x >> 5);
  int lane = threadIdx.x & 31;
  int ncol = lane & 15, hlf = lane >> 4, mrow = lane & 15;
  const int nTiles64 = HIDDEN / 64;  // 16
  int tm = waveId / nTiles64, tn = waveId % nTiles64;
  int m0 = tm * 16, n0 = tn * 64;
  if (m0 >= Mrows) return;

  const __bf16* arow = X + (size_t)(m0 + mrow) * HIDDEN;
  v8f c[4] = {};
#pragma unroll 2
  for (int ch = 0; ch < HIDDEN / 32; ++ch) {
    int kbase = ch * 32;
    v16bf a = load_a_from_bf16(arow, kbase, hlf);
#pragma unroll
    for (int q = 0; q < 4; ++q) {
      v16bf b = load_b_from_bf16t(Wt, HIDDEN, kbase, n0 + q * 16, lane);
      c[q] = __builtin_amdgcn_wmma_f32_16x16x32_bf16(false, a, false, b,
                                                     (short)0, c[q], false,
                                                     false);
    }
  }

#pragma unroll
  for (int q = 0; q < 4; ++q) {
#pragma unroll
    for (int r = 0; r < 8; ++r) {
      int M = r + 8 * hlf;
      int row = m0 + M;
      int j = n0 + q * 16 + ncol;
      float v = c[q][r] + bias[j];
      size_t idx;
      if (mode == 0) {
        idx = (size_t)row * HIDDEN + j;
      } else {
        int bb = row / rowsPerB;
        int s = row - bb * rowsPerB;
        int hh = j >> 6, d = j & 63;
        if (mode == 1) {
          v += seg_emb[(size_t)seg_ids[row] * HIDDEN + j];
          idx = (((size_t)bb * HEADS + hh) * SS + s) * HEAD_DIM + d;
        } else {
          idx = (((size_t)bb * HEADS + hh) * HEAD_DIM + d) * SS + s;
        }
      }
      out[idx] = (__bf16)v;
    }
  }
}

// -------- attention core: one block per (b, h, 16-query tile) ---------------

__global__ void __launch_bounds__(128)
attn_kernel(const __bf16* __restrict__ Qbf, const __bf16* __restrict__ Kbf,
            const __bf16* __restrict__ Vbf, const int* __restrict__ amask,
            float* __restrict__ prov, __bf16* __restrict__ attBf) {
  extern __shared__ float panel[];  // [16][4096] fp32 scores -> exp weights
  __shared__ float rowmax[16], rowinv[16];
  __shared__ float pmax[16][8], psum[16][8];

  int blk = blockIdx.x;            // b*HEADS*32 + h*32 + ttile
  int ttile = blk & 31;
  int bh = blk >> 5;
  int h = bh & (HEADS - 1);
  int b = bh >> 4;
  int t0 = ttile * 16;

  int tid = threadIdx.x;
  int lane = tid & 31, wv = tid >> 5;
  int mrow = lane & 15, hlf = lane >> 4, ncol = mrow;

  // Q fragments for this 16-row tile (k = 0..31 and 32..63 of head_dim)
  const __bf16* qrow =
      Qbf + (size_t)(b * TT + t0 + mrow) * HIDDEN + h * HEAD_DIM;
  v16bf aq0 = load_a_from_bf16(qrow, 0, hlf);
  v16bf aq1 = load_a_from_bf16(qrow, 32, hlf);

  // ---- phase 1: scores = Q K^T / 8, masked, into LDS panel ----
  for (int tile = wv; tile < SS / 16; tile += 4) {
    int s0 = tile * 16;
    const __bf16* krow = Kbf + ((size_t)bh * SS + s0 + ncol) * HEAD_DIM;
    if (tile + 4 < SS / 16)
      __builtin_prefetch(krow + 64 * HEAD_DIM, 0, 0);  // global_prefetch_b8
    v16bf bk0 = *(const v16bf*)(krow + hlf * 16);
    v16bf bk1 = *(const v16bf*)(krow + 32 + hlf * 16);
    v8f c = {};
    c = __builtin_amdgcn_wmma_f32_16x16x32_bf16(false, aq0, false, bk0,
                                                (short)0, c, false, false);
    c = __builtin_amdgcn_wmma_f32_16x16x32_bf16(false, aq1, false, bk1,
                                                (short)0, c, false, false);
    int am = amask[b * SS + s0 + ncol];
#pragma unroll
    for (int r = 0; r < 8; ++r) {
      float v = am ? c[r] * 0.125f : -1.0e9f;
      panel[(r + 8 * hlf) * SS + s0 + ncol] = v;
    }
  }
  __syncthreads();

  // ---- phase 2: exact softmax over each 4096-wide row (8 threads/row) ----
  int row = tid >> 3, seg = tid & 7;
  float* prow = panel + row * SS + seg * 512;
  float mx = -3.0e38f;
  for (int i = 0; i < 512; ++i) mx = fmaxf(mx, prow[i]);
  pmax[row][seg] = mx;
  __syncthreads();
  if (seg == 0) {
    float m = pmax[row][0];
#pragma unroll
    for (int i = 1; i < 8; ++i) m = fmaxf(m, pmax[row][i]);
    rowmax[row] = m;
  }
  __syncthreads();
  float m = rowmax[row];
  float sm = 0.0f;
  for (int i = 0; i < 512; ++i) {
    float e = __expf(prow[i] - m);
    prow[i] = e;
    sm += e;
  }
  psum[row][seg] = sm;
  __syncthreads();
  if (seg == 0) {
    float s = 0.0f;
#pragma unroll
    for (int i = 0; i < 8; ++i) s += psum[row][i];
    rowinv[row] = 1.0f / s;
  }
  __syncthreads();

  // ---- phase 3: provenance += weights / HEADS (fp32 HW atomics) ----
  float sc = rowinv[row] * (1.0f / (float)HEADS);
  float* pv = prov + ((size_t)b * TT + t0 + row) * SS + seg * 512;
  for (int i = 0; i < 512; ++i) unsafeAtomicAdd(pv + i, prow[i] * sc);

  // ---- phase 4: attended = weights @ V ; wave wv owns d-range wv*16 ----
  int d0 = wv * 16;
  v8f acc = {};
  const __bf16* vrow = Vbf + ((size_t)bh * HEAD_DIM + d0 + ncol) * SS;
  const float* wrow = panel + mrow * SS;
  for (int ch = 0; ch < SS / 32; ++ch) {
    int k0 = ch * 32;
    v16bf aw;
#pragma unroll
    for (int j = 0; j < 16; ++j) {
      aw[j] = (__bf16)wrow[k0 + hlf * 16 + j];
    }
    v16bf bv = *(const v16bf*)(vrow + k0 + hlf * 16);
    acc = __builtin_amdgcn_wmma_f32_16x16x32_bf16(false, aw, false, bv,
                                                  (short)0, acc, false, false);
  }
#pragma unroll
  for (int r = 0; r < 8; ++r) {
    int M = r + 8 * hlf;
    float v = acc[r] * rowinv[M];
    attBf[((size_t)(b * TT) + t0 + M) * HIDDEN + h * HEAD_DIM + d0 + ncol] =
        (__bf16)v;
  }
}

// -------- output projection: d_out[:2M] = attended @ Wo + bo (fp32 out) -----
// One wave computes a 16x64 tile from bf16 attended and bf16 Wo^T.

__global__ void __launch_bounds__(128)
outproj_kernel(const __bf16* __restrict__ Abf, const __bf16* __restrict__ WoT,
               const float* __restrict__ bo, float* __restrict__ out) {
  int waveId = blockIdx.x * 4 + (threadIdx.x >> 5);
  int lane = threadIdx.x & 31;
  int ncol = lane & 15, hlf = lane >> 4, mrow = lane & 15;
  const int nTiles64 = HIDDEN / 64;
  int tm = waveId / nTiles64, tn = waveId % nTiles64;
  int m0 = tm * 16, n0 = tn * 64;
  if (m0 >= BB * TT) return;

  const __bf16* arow = Abf + (size_t)(m0 + mrow) * HIDDEN;
  v8f c[4] = {};
#pragma unroll 2
  for (int ch = 0; ch < HIDDEN / 32; ++ch) {
    int kbase = ch * 32;
    v16bf a = load_a_from_bf16(arow, kbase, hlf);
#pragma unroll
    for (int q = 0; q < 4; ++q) {
      v16bf b = load_b_from_bf16t(WoT, HIDDEN, kbase, n0 + q * 16, lane);
      c[q] = __builtin_amdgcn_wmma_f32_16x16x32_bf16(false, a, false, b,
                                                     (short)0, c[q], false,
                                                     false);
    }
  }
#pragma unroll
  for (int q = 0; q < 4; ++q)
#pragma unroll
    for (int r = 0; r < 8; ++r) {
      int M = r + 8 * hlf;
      int j = n0 + q * 16 + ncol;
      out[(size_t)(m0 + M) * HIDDEN + j] = c[q][r] + bo[j];
    }
}

// ---------------------------------------------------------------------------

extern "C" void kernel_launch(void* const* d_in, const int* in_sizes, int n_in,
                              void* d_out, int out_size, void* d_ws,
                              size_t ws_size, hipStream_t stream) {
  const float* query    = (const float*)d_in[0];
  const float* key      = (const float*)d_in[1];
  const float* value    = (const float*)d_in[2];
  const int*   amask    = (const int*)d_in[3];
  const int*   seg_ids  = (const int*)d_in[4];
  const float* Wq       = (const float*)d_in[5];
  const float* bq       = (const float*)d_in[6];
  const float* Wk       = (const float*)d_in[7];
  const float* bk       = (const float*)d_in[8];
  const float* Wv       = (const float*)d_in[9];
  const float* bv       = (const float*)d_in[10];
  const float* Wo       = (const float*)d_in[11];
  const float* bo       = (const float*)d_in[12];
  const float* seg_emb  = (const float*)d_in[13];

  char* ws = (char*)d_ws;
  const size_t MB = (size_t)1 << 20;
  // bf16 transposed weights: 2 MiB each
  __bf16* WqT = (__bf16*)(ws + 0 * MB);
  __bf16* WkT = (__bf16*)(ws + 2 * MB);
  __bf16* WvT = (__bf16*)(ws + 4 * MB);
  __bf16* WoT = (__bf16*)(ws + 6 * MB);
  // bf16 activations (Qin buffer is reused for the attended intermediate)
  __bf16* Qin = (__bf16*)(ws + 8 * MB);    //  4 MiB (aliased as Abf later)
  __bf16* Abf = Qin;
  __bf16* Kin = (__bf16*)(ws + 12 * MB);   // 32 MiB
  __bf16* Vin = (__bf16*)(ws + 44 * MB);   // 32 MiB
  // bf16 projected tensors
  __bf16* Qbf = (__bf16*)(ws + 76 * MB);   //  4 MiB
  __bf16* Kbf = (__bf16*)(ws + 80 * MB);   // 32 MiB
  __bf16* Vbf = (__bf16*)(ws + 112 * MB);  // 32 MiB  (144 MiB total)

  float* out  = (float*)d_out;
  float* prov = out + (size_t)BB * TT * HIDDEN;  // provenance region

  // 1) zero provenance (accumulated across heads via atomics)
  {
    int n = BB * TT * SS;
    zero_f32_kernel<<<(n + 255) / 256, 256, 0, stream>>>(prov, n);
  }
  // 2) convert activations to bf16; transpose+convert weights to bf16 [n][k]
  {
    int nq = BB * TT * HIDDEN, nkv = BB * SS * HIDDEN;
    cvt_f32_to_bf16_kernel<<<(nq + 255) / 256, 256, 0, stream>>>(query, Qin,
                                                                 nq);
    cvt_f32_to_bf16_kernel<<<(nkv + 255) / 256, 256, 0, stream>>>(key, Kin,
                                                                  nkv);
    cvt_f32_to_bf16_kernel<<<(nkv + 255) / 256, 256, 0, stream>>>(value, Vin,
                                                                  nkv);
    transpose_cvt_kernel<<<1024, 256, 0, stream>>>(Wq, WqT);
    transpose_cvt_kernel<<<1024, 256, 0, stream>>>(Wk, WkT);
    transpose_cvt_kernel<<<1024, 256, 0, stream>>>(Wv, WvT);
    transpose_cvt_kernel<<<1024, 256, 0, stream>>>(Wo, WoT);
  }
  // 3) projections to bf16 workspace (16x64 tiles per wave)
  {
    int wavesQ = (BB * TT / 16) * (HIDDEN / 64);   // 2048
    proj_kernel<<<wavesQ / 4, 128, 0, stream>>>(Qin, WqT, bq, nullptr,
                                                nullptr, Qbf, BB * TT, TT, 0);
    int wavesKV = (BB * SS / 16) * (HIDDEN / 64);  // 16384
    proj_kernel<<<wavesKV / 4, 128, 0, stream>>>(Kin, WkT, bk, seg_ids,
                                                 seg_emb, Kbf, BB * SS, SS, 1);
    proj_kernel<<<wavesKV / 4, 128, 0, stream>>>(Vin, WvT, bv, nullptr,
                                                 nullptr, Vbf, BB * SS, SS, 2);
  }
  // 4) attention core: 256KB dynamic LDS score panel (320KB/WGP on MI455X)
  {
    int blocks = BB * HEADS * (TT / 16);  // 2048
    size_t shmem = (size_t)16 * SS * sizeof(float);
    attn_kernel<<<blocks, 128, shmem, stream>>>(Qbf, Kbf, Vbf, amask, prov,
                                                Abf);
  }
  // 5) output projection straight into d_out
  {
    int waves = (BB * TT / 16) * (HIDDEN / 64);  // 2048
    outproj_kernel<<<waves / 4, 128, 0, stream>>>(Abf, WoT, bo, out);
  }
}
